// Darnn_selfattention_47571057770989
// MI455X (gfx1250) — matrix operand order repaired and checked by reading, not compiled, not run on hardware
//
#include <hip/hip_runtime.h>

// DA-RNN forward for MI455X (gfx1250, wave32, WMMA f16 16x16x32, f32 accum).
//
// Input flattening assumption (jax tree_leaves: dict keys sorted, tuples in order):
//  0 dec_attn1.W[128,384]  1 dec_attn1.b[128]  2 dec_attn2.W[1,128]  3 dec_attn2.b[1]
//  4 dec_fc.W[1,129]       5 dec_fc.b[1]
//  6 dec_lstm.Whh[512,128] 7 dec_lstm.Wih[512,1] 8 dec_lstm.bhh[512] 9 dec_lstm.bih[512]
// 10 dec_sa.Wk 11 dec_sa.Wq 12 dec_sa.Wv [128,128]  13 bk 14 bq 15 bv [128]
// 16 enc_attn1.W[64,320]  17 enc_attn1.b[64] 18 enc_attn2.W[1,64] 19 enc_attn2.b[1]
// 20 enc_lstm.Whh[512,128] 21 enc_lstm.Wih[512,64] 22 bhh[512] 23 bih[512]
// 24 enc_sa.Wk 25 Wq 26 Wv [64,64]  27 bk 28 bq 29 bv [64]
// 30 last_fc.W[1,128] 31 last_fc.b[1]
// 32 top_sa.Wk 33 Wq 34 Wv [128,128]  35 bk 36 bq 37 bv [128]
// 38 x[2048,64,64]  39 y[2048,64]

typedef _Float16 half_t;
typedef __attribute__((ext_vector_type(16))) _Float16 v16h;
typedef __attribute__((ext_vector_type(8)))  float    v8f;

#define DEV static __device__ __forceinline__

DEV v8f wmma32(v16h a, v16h b, v8f c) {
  return __builtin_amdgcn_wmma_f32_16x16x32_f16(false, a, false, b, (short)0, c,
                                                false, false);
}

// ---- fragment loaders (wave32; contiguous k = 16*g + i in both A and B) ----
// A 16x32 f16: lane m=lane&15, g=lane>>4; element i -> K = 16*g + i
template <typename T>
DEV v16h fragA(const T* src, int ld) {
  const int lane = threadIdx.x & 31;
  const int m = lane & 15, g = lane >> 4;
  v16h a;
#pragma unroll
  for (int i = 0; i < 16; ++i) {
    const int k = (g << 4) + i;
    a[i] = (_Float16)src[m * ld + k];
  }
  return a;
}
// B 32x16 f16 from row-major [K][N]: lane n=lane&15, g=lane>>4; K = 16*g + i
template <typename T>
DEV v16h fragB(const T* src, int ld) {
  const int lane = threadIdx.x & 31;
  const int n = lane & 15, g = lane >> 4;
  v16h b;
#pragma unroll
  for (int i = 0; i < 16; ++i) b[i] = (_Float16)src[((g << 4) + i) * ld + n];
  return b;
}
// B 32x16 f16 from row-major [N][K] (i.e. transposed source)
template <typename T>
DEV v16h fragBT(const T* src, int ld) {
  const int lane = threadIdx.x & 31;
  const int n = lane & 15, g = lane >> 4;
  v16h b;
#pragma unroll
  for (int i = 0; i < 16; ++i) b[i] = (_Float16)src[n * ld + (g << 4) + i];
  return b;
}
// Packed fragment (A or B tile pre-swizzled): 32 lanes x 16 contiguous halves
// -> one 32-byte vector load per lane (global_load_b128 x2 / ds_load_b128 x2).
DEV v16h fragP(const half_t* base) {
  const int lane = threadIdx.x & 31;
  const half_t* p = base + lane * 16;
  return *(const v16h*)__builtin_assume_aligned(p, 32);
}
// A-fragment packed element map: idx i of lane -> k = 16*g + i
DEV int packA_k(int lane, int i) {
  return (((lane >> 4) << 4) + i);
}
// C/D: VGPR r -> row = r + 8*g, col = lane&15

DEV float sigmoidf_(float x) { return 1.f / (1.f + __expf(-x)); }

// ============================ prep kernels =================================
__global__ void zero_kernel(float* p, int n) {
  int i = blockIdx.x * 256 + threadIdx.x;
  if (i < n) p[i] = 0.f;
}

// Pack weight [N_out][K_in] f32 -> f16 B-fragment tiles.
// Tile t = (k/32)*(N/16) + (n/16); out[t*512 + lane*16 + i] = in[n][kOff + k]
// with n = nTile*16 + (lane&15), k = kTile*32 + 16*(lane>>4) + i.
__global__ void tpack_kernel(const float* __restrict__ in, half_t* __restrict__ out,
                             int K, int N, int ldin, int kOff) {
  int idx = blockIdx.x * 256 + threadIdx.x;
  if (idx < K * N) {
    const int tile = idx >> 9, r = idx & 511;
    const int lane = r >> 4, i = r & 15;
    const int ntiles = N >> 4;
    const int kt = tile / ntiles, nt = tile - kt * ntiles;
    const int n = nt * 16 + (lane & 15);
    const int k = kt * 32 + ((lane >> 4) << 4) + i;
    out[idx] = (_Float16)in[n * ldin + kOff + k];
  }
}

// =================== generic batched GEMM: C[b] = A[b](64xK) @ Bp(KxN) + bias
// Bp is fragment-packed; A staged into LDS in A-fragment-packed order.
template <typename AT>
__global__ __launch_bounds__(256) void gemm64_kernel(
    const AT* __restrict__ A, const half_t* __restrict__ Bp,
    const float* __restrict__ bias, float* __restrict__ Cf,
    half_t* __restrict__ Ch, int N, int K) {
  __shared__ __attribute__((aligned(32))) half_t sA[64 * 128];  // K <= 128
  const int b = blockIdx.x, tid = threadIdx.x, wave = tid >> 5;
  const int ktiles = K >> 5;
  const AT* Ab = A + (size_t)b * 64 * K;
  // packed stage: tile = mt*ktiles + kt
  for (int idx = tid; idx < 64 * K; idx += 256) {
    const int tile = idx >> 9, r = idx & 511;
    const int lane = r >> 4, i = r & 15;
    const int mt = tile / ktiles, kt = tile - mt * ktiles;
    const int m = mt * 16 + (lane & 15);
    const int k = kt * 32 + packA_k(lane, i);
    sA[idx] = (_Float16)Ab[m * K + k];
  }
  __syncthreads();
  const int ntiles16 = N >> 4;
  const int ntiles = 4 * ntiles16;
  for (int tile = wave; tile < ntiles; tile += 8) {
    const int mt = tile & 3, nt = tile >> 2;
    v8f acc = {};
    for (int kc = 0; kc < K; kc += 32) {
      v16h a = fragP(sA + (mt * ktiles + (kc >> 5)) * 512);
      v16h bb = fragP(Bp + ((kc >> 5) * ntiles16 + nt) * 512);
      acc = wmma32(a, bb, acc);
    }
    const int lane = tid & 31, n = lane & 15, g = lane >> 4;
    const float bvv = bias ? bias[nt * 16 + n] : 0.f;
#pragma unroll
    for (int r = 0; r < 8; ++r) {
      const int row = mt * 16 + r + 8 * g;
      const float v = acc[r] + bvv;
      if (Cf) Cf[(size_t)b * 64 * N + row * N + nt * 16 + n] = v;
      if (Ch) Ch[(size_t)b * 64 * N + row * N + nt * 16 + n] = (_Float16)v;
    }
  }
}

// ======================= encoder self-attention (64x64, per batch) =========
__global__ __launch_bounds__(256) void enc_sa_kernel(
    const float* __restrict__ x, const half_t* __restrict__ WqP,
    const half_t* __restrict__ WkP, const half_t* __restrict__ WvP,
    const float* __restrict__ bq, const float* __restrict__ bk,
    const float* __restrict__ bv, float* __restrict__ Xsa) {
  __shared__ half_t sX[64 * 68];
  __shared__ half_t sQ[64 * 68];  // Q, later P
  __shared__ half_t sK[64 * 68];
  __shared__ half_t sV[64 * 68];
  __shared__ float sS[64 * 68];
  const int b = blockIdx.x, tid = threadIdx.x, wave = tid >> 5;
  const int lane = tid & 31, ln = lane & 15, g = lane >> 4;
  const float* xb = x + (size_t)b * 64 * 64;
  // sX[i][t] = x[b][t][i]  (transpose on load)
  for (int idx = tid; idx < 64 * 64; idx += 256) {
    const int i = idx & 63, t = idx >> 6;
    sX[i * 68 + t] = (_Float16)xb[t * 64 + i];
  }
  __syncthreads();
  for (int which = 0; which < 3; ++which) {
    const half_t* Wp = which == 0 ? WqP : which == 1 ? WkP : WvP;
    const float* bias = which == 0 ? bq : which == 1 ? bk : bv;
    half_t* out = which == 0 ? sQ : which == 1 ? sK : sV;
    for (int tile = wave; tile < 16; tile += 8) {
      const int mt = tile & 3, nt = tile >> 2;
      v8f acc = {};
#pragma unroll
      for (int kc = 0; kc < 64; kc += 32) {
        v16h a = fragA(sX + mt * 16 * 68 + kc, 68);
        v16h bb = fragP(Wp + ((kc >> 5) * 4 + nt) * 512);
        acc = wmma32(a, bb, acc);
      }
      const float bvv = bias[nt * 16 + ln];
#pragma unroll
      for (int r = 0; r < 8; ++r)
        out[(mt * 16 + r + 8 * g) * 68 + nt * 16 + ln] = (_Float16)(acc[r] + bvv);
    }
  }
  __syncthreads();
  // S = Q K^T / sqrt(64)
  for (int tile = wave; tile < 16; tile += 8) {
    const int mt = tile & 3, nt = tile >> 2;
    v8f acc = {};
#pragma unroll
    for (int kc = 0; kc < 64; kc += 32) {
      v16h a = fragA(sQ + mt * 16 * 68 + kc, 68);
      v16h bb = fragBT(sK + nt * 16 * 68 + kc, 68);
      acc = wmma32(a, bb, acc);
    }
#pragma unroll
    for (int r = 0; r < 8; ++r)
      sS[(mt * 16 + r + 8 * g) * 68 + nt * 16 + ln] = acc[r] * 0.125f;
  }
  __syncthreads();
  if (tid < 64) {
    float mx = -1e30f;
    for (int j = 0; j < 64; ++j) mx = fmaxf(mx, sS[tid * 68 + j]);
    float sum = 0.f;
    for (int j = 0; j < 64; ++j) {
      float e = __expf(sS[tid * 68 + j] - mx);
      sS[tid * 68 + j] = e;
      sum += e;
    }
    const float inv = 1.f / sum;
    for (int j = 0; j < 64; ++j) sQ[tid * 68 + j] = (_Float16)(sS[tid * 68 + j] * inv);
  }
  __syncthreads();
  float* ob = Xsa + (size_t)b * 64 * 64;
  for (int tile = wave; tile < 16; tile += 8) {
    const int mt = tile & 3, nt = tile >> 2;
    v8f acc = {};
#pragma unroll
    for (int kc = 0; kc < 64; kc += 32) {
      v16h a = fragA(sQ + mt * 16 * 68 + kc, 68);
      v16h bb = fragB(sV + kc * 68 + nt * 16, 68);
      acc = wmma32(a, bb, acc);
    }
#pragma unroll
    for (int r = 0; r < 8; ++r)
      ob[(mt * 16 + r + 8 * g) * 64 + nt * 16 + ln] = acc[r];
  }
}

// ======================= encoder scan step (fused) =========================
__global__ __launch_bounds__(256) void enc_step_kernel(
    int t, const float* __restrict__ Xsa, const float* __restrict__ XsaPart,
    const half_t* __restrict__ W1hcP,  // packed [256][64]
    const float* __restrict__ W2, const float* __restrict__ b2,
    const half_t* __restrict__ WihP,  // packed [64][512]
    const half_t* __restrict__ WhhP,  // packed [128][512]
    const float* __restrict__ bih, const float* __restrict__ bhh,
    float* __restrict__ hS, float* __restrict__ cS, half_t* __restrict__ Xenc) {
  __shared__ __attribute__((aligned(32))) half_t sHC[16 * 256];  // A-packed, 8 k-tiles
  __shared__ __attribute__((aligned(32))) half_t sXt[16 * 64];   // A-packed, 2 k-tiles
  __shared__ float sE1[16 * 64];
  __shared__ float sAl[16 * 64];
  __shared__ float sG[16 * 512];
  const int tid = threadIdx.x, wave = tid >> 5;
  const int lane = tid & 31, ln = lane & 15, g = lane >> 4;
  const int b0 = blockIdx.x * 16;
  // stage h|c (cols 0..127 = h, 128..255 = c) in A-fragment-packed order
  for (int idx = tid; idx < 16 * 256; idx += 256) {
    const int kt = idx >> 9, r = idx & 511;
    const int l = r >> 4, i = r & 15;
    const int m = l & 15;
    const int k = kt * 32 + packA_k(l, i);
    const float v = (k < 128) ? hS[(b0 + m) * 128 + k] : cS[(b0 + m) * 128 + (k - 128)];
    sHC[idx] = (_Float16)v;
  }
  __syncthreads();
  // hc-part of attention MLP: [16,256] @ [256,64]
  if (wave < 4) {
    const int nt = wave;
    v8f acc = {};
    for (int kc = 0; kc < 256; kc += 32) {
      v16h a = fragP(sHC + (kc >> 5) * 512);
      v16h bb = fragP(W1hcP + ((kc >> 5) * 4 + nt) * 512);
      acc = wmma32(a, bb, acc);
    }
#pragma unroll
    for (int r = 0; r < 8; ++r) sE1[(r + 8 * g) * 64 + nt * 16 + ln] = acc[r];
  }
  __syncthreads();
  // energies e[bi][n]
  for (int idx = tid; idx < 16 * 64; idx += 256) {
    const int bi = idx >> 6, n = idx & 63;
    const float* xp = XsaPart + ((size_t)(b0 + bi) * 64 + n) * 64;
    float acc = b2[0];
    for (int j = 0; j < 64; ++j) acc += tanhf(sE1[bi * 64 + j] + xp[j]) * W2[j];
    sAl[idx] = acc;
  }
  __syncthreads();
  if (tid < 16) {
    float mx = -1e30f;
    for (int n = 0; n < 64; ++n) mx = fmaxf(mx, sAl[tid * 64 + n]);
    float sum = 0.f;
    for (int n = 0; n < 64; ++n) {
      float e = __expf(sAl[tid * 64 + n] - mx);
      sAl[tid * 64 + n] = e;
      sum += e;
    }
    const float inv = 1.f / sum;
    for (int n = 0; n < 64; ++n) sAl[tid * 64 + n] *= inv;
  }
  __syncthreads();
  // x_tilde = alpha * x_t, staged in A-fragment-packed order
  for (int idx = tid; idx < 16 * 64; idx += 256) {
    const int kt = idx >> 9, r = idx & 511;
    const int l = r >> 4, i = r & 15;
    const int bi = l & 15;
    const int n = kt * 32 + packA_k(l, i);
    const float xt = Xsa[((size_t)(b0 + bi) * 64 + n) * 64 + t];
    sXt[idx] = (_Float16)(sAl[bi * 64 + n] * xt);
  }
  __syncthreads();
  // LSTM gates: [16,64]@[64,512] + [16,128]@[128,512]
  for (int tile = wave; tile < 32; tile += 8) {
    const int nt = tile;
    v8f acc = {};
#pragma unroll
    for (int kc = 0; kc < 64; kc += 32) {
      v16h a = fragP(sXt + (kc >> 5) * 512);
      v16h bb = fragP(WihP + ((kc >> 5) * 32 + nt) * 512);
      acc = wmma32(a, bb, acc);
    }
#pragma unroll
    for (int kc = 0; kc < 128; kc += 32) {
      v16h a = fragP(sHC + (kc >> 5) * 512);  // h part = k-tiles 0..3
      v16h bb = fragP(WhhP + ((kc >> 5) * 32 + nt) * 512);
      acc = wmma32(a, bb, acc);
    }
#pragma unroll
    for (int r = 0; r < 8; ++r) sG[(r + 8 * g) * 512 + nt * 16 + ln] = acc[r];
  }
  __syncthreads();
  for (int idx = tid; idx < 16 * 128; idx += 256) {
    const int bi = idx >> 7, k = idx & 127;
    const float gi = sG[bi * 512 + k] + bih[k] + bhh[k];
    const float gf = sG[bi * 512 + 128 + k] + bih[128 + k] + bhh[128 + k];
    const float gg = sG[bi * 512 + 256 + k] + bih[256 + k] + bhh[256 + k];
    const float go = sG[bi * 512 + 384 + k] + bih[384 + k] + bhh[384 + k];
    const float co = cS[(b0 + bi) * 128 + k];
    const float cn = sigmoidf_(gf) * co + sigmoidf_(gi) * tanhf(gg);
    const float hn = sigmoidf_(go) * tanhf(cn);
    cS[(b0 + bi) * 128 + k] = cn;
    hS[(b0 + bi) * 128 + k] = hn;
    Xenc[((size_t)(b0 + bi) * 64 + t) * 128 + k] = (_Float16)hn;
  }
}

// ======================= decoder self-attention (64x128, per batch) ========
__global__ __launch_bounds__(256) void dec_sa_kernel(
    const half_t* __restrict__ Xenc, const half_t* __restrict__ WqP,
    const half_t* __restrict__ WkP, const half_t* __restrict__ WvP,
    const float* __restrict__ bq, const float* __restrict__ bk,
    const float* __restrict__ bv, half_t* __restrict__ Xe) {
  __shared__ half_t sA[64 * 128];  // Q then P
  __shared__ half_t sB[64 * 128];  // K then V
  __shared__ float sS[64 * 64];
  const int b = blockIdx.x, tid = threadIdx.x, wave = tid >> 5;
  const int lane = tid & 31, ln = lane & 15, g = lane >> 4;
  const half_t* xb = Xenc + (size_t)b * 64 * 128;
  for (int which = 0; which < 2; ++which) {
    const half_t* Wp = which ? WkP : WqP;
    const float* bias = which ? bk : bq;
    half_t* out = which ? sB : sA;
    for (int tile = wave; tile < 32; tile += 8) {
      const int mt = tile & 3, nt = tile >> 2;
      v8f acc = {};
#pragma unroll
      for (int kc = 0; kc < 128; kc += 32) {
        v16h a = fragA(xb + mt * 16 * 128 + kc, 128);
        v16h bb = fragP(Wp + ((kc >> 5) * 8 + nt) * 512);
        acc = wmma32(a, bb, acc);
      }
      const float bvv = bias[nt * 16 + ln];
#pragma unroll
      for (int r = 0; r < 8; ++r)
        out[(mt * 16 + r + 8 * g) * 128 + nt * 16 + ln] = (_Float16)(acc[r] + bvv);
    }
  }
  __syncthreads();
  const float scale = 0.088388347648318447f;  // 1/sqrt(128)
  for (int tile = wave; tile < 16; tile += 8) {
    const int mt = tile & 3, nt = tile >> 2;
    v8f acc = {};
#pragma unroll
    for (int kc = 0; kc < 128; kc += 32) {
      v16h a = fragA(sA + mt * 16 * 128 + kc, 128);
      v16h bb = fragBT(sB + nt * 16 * 128 + kc, 128);
      acc = wmma32(a, bb, acc);
    }
#pragma unroll
    for (int r = 0; r < 8; ++r)
      sS[(mt * 16 + r + 8 * g) * 64 + nt * 16 + ln] = acc[r] * scale;
  }
  __syncthreads();
  if (tid < 64) {
    float mx = -1e30f;
    for (int j = 0; j < 64; ++j) mx = fmaxf(mx, sS[tid * 64 + j]);
    float sum = 0.f;
    for (int j = 0; j < 64; ++j) {
      float e = __expf(sS[tid * 64 + j] - mx);
      sS[tid * 64 + j] = e;
      sum += e;
    }
    const float inv = 1.f / sum;
    for (int j = 0; j < 64; ++j) sA[tid * 128 + j] = (_Float16)(sS[tid * 64 + j] * inv);
  }
  __syncthreads();
  // V into sB
  for (int tile = wave; tile < 32; tile += 8) {
    const int mt = tile & 3, nt = tile >> 2;
    v8f acc = {};
#pragma unroll
    for (int kc = 0; kc < 128; kc += 32) {
      v16h a = fragA(xb + mt * 16 * 128 + kc, 128);
      v16h bb = fragP(WvP + ((kc >> 5) * 8 + nt) * 512);
      acc = wmma32(a, bb, acc);
    }
    const float bvv = bv[nt * 16 + ln];
#pragma unroll
    for (int r = 0; r < 8; ++r)
      sB[(mt * 16 + r + 8 * g) * 128 + nt * 16 + ln] = (_Float16)(acc[r] + bvv);
  }
  __syncthreads();
  half_t* ob = Xe + (size_t)b * 64 * 128;
  for (int tile = wave; tile < 32; tile += 8) {
    const int mt = tile & 3, nt = tile >> 2;
    v8f acc = {};
#pragma unroll
    for (int kc = 0; kc < 64; kc += 32) {
      v16h a = fragA(sA + mt * 16 * 128 + kc, 128);
      v16h bb = fragB(sB + kc * 128 + nt * 16, 128);
      acc = wmma32(a, bb, acc);
    }
#pragma unroll
    for (int r = 0; r < 8; ++r)
      ob[(mt * 16 + r + 8 * g) * 128 + nt * 16 + ln] = (_Float16)acc[r];
  }
}

// ======================= decoder scan step (fused) =========================
__global__ __launch_bounds__(256) void dec_step_kernel(
    int t, const float* __restrict__ y, const half_t* __restrict__ Xe,
    const float* __restrict__ XePart,
    const half_t* __restrict__ Wa1dcP,  // packed [256][128]
    const float* __restrict__ Wa2, const float* __restrict__ ba2,
    const float* __restrict__ Wf, const float* __restrict__ bf,
    const float* __restrict__ Wih,    // [512]
    const half_t* __restrict__ WhhP,  // packed [128][512]
    const float* __restrict__ bih, const float* __restrict__ bhh,
    float* __restrict__ dS, float* __restrict__ cS) {
  __shared__ __attribute__((aligned(32))) half_t sDC[16 * 256];  // A-packed
  __shared__ float sE1[16 * 128];
  __shared__ float sBe[16 * 64];
  __shared__ float sCtx[16 * 128];
  __shared__ float sYt[16];
  __shared__ float sG[16 * 512];
  const int tid = threadIdx.x, wave = tid >> 5;
  const int lane = tid & 31, ln = lane & 15, g = lane >> 4;
  const int b0 = blockIdx.x * 16;
  for (int idx = tid; idx < 16 * 256; idx += 256) {
    const int kt = idx >> 9, r = idx & 511;
    const int l = r >> 4, i = r & 15;
    const int m = l & 15;
    const int k = kt * 32 + packA_k(l, i);
    const float v = (k < 128) ? dS[(b0 + m) * 128 + k] : cS[(b0 + m) * 128 + (k - 128)];
    sDC[idx] = (_Float16)v;
  }
  __syncthreads();
  {  // dc-part of attention MLP: [16,256]@[256,128], 8 N-tiles, 1 per wave
    const int nt = wave;
    v8f acc = {};
    for (int kc = 0; kc < 256; kc += 32) {
      v16h a = fragP(sDC + (kc >> 5) * 512);
      v16h bb = fragP(Wa1dcP + ((kc >> 5) * 8 + nt) * 512);
      acc = wmma32(a, bb, acc);
    }
#pragma unroll
    for (int r = 0; r < 8; ++r) sE1[(r + 8 * g) * 128 + nt * 16 + ln] = acc[r];
  }
  __syncthreads();
  for (int idx = tid; idx < 16 * 64; idx += 256) {
    const int bi = idx >> 6, tt = idx & 63;
    const float* xp = XePart + ((size_t)(b0 + bi) * 64 + tt) * 128;
    float acc = ba2[0];
    for (int j = 0; j < 128; ++j) acc += tanhf(sE1[bi * 128 + j] + xp[j]) * Wa2[j];
    sBe[idx] = acc;
  }
  __syncthreads();
  if (tid < 16) {
    float mx = -1e30f;
    for (int n = 0; n < 64; ++n) mx = fmaxf(mx, sBe[tid * 64 + n]);
    float sum = 0.f;
    for (int n = 0; n < 64; ++n) {
      float e = __expf(sBe[tid * 64 + n] - mx);
      sBe[tid * 64 + n] = e;
      sum += e;
    }
    const float inv = 1.f / sum;
    for (int n = 0; n < 64; ++n) sBe[tid * 64 + n] *= inv;
  }
  __syncthreads();
  // context = beta @ Xe
  for (int idx = tid; idx < 16 * 128; idx += 256) {
    const int bi = idx >> 7, hh = idx & 127;
    const half_t* xeb = Xe + (size_t)(b0 + bi) * 64 * 128 + hh;
    float acc = 0.f;
    for (int tt = 0; tt < 64; ++tt) acc += sBe[bi * 64 + tt] * (float)xeb[tt * 128];
    sCtx[idx] = acc;
  }
  __syncthreads();
  if (tid < 16) {
    float acc = bf[0] + Wf[128] * y[(b0 + tid) * 64 + t];
    for (int j = 0; j < 128; ++j) acc += Wf[j] * sCtx[tid * 128 + j];
    sYt[tid] = acc;
  }
  __syncthreads();
  // gates: d @ WhhT (y_tilde*Wih added in epilogue)
  for (int tile = wave; tile < 32; tile += 8) {
    const int nt = tile;
    v8f acc = {};
#pragma unroll
    for (int kc = 0; kc < 128; kc += 32) {
      v16h a = fragP(sDC + (kc >> 5) * 512);  // d part = k-tiles 0..3
      v16h bb = fragP(WhhP + ((kc >> 5) * 32 + nt) * 512);
      acc = wmma32(a, bb, acc);
    }
#pragma unroll
    for (int r = 0; r < 8; ++r) sG[(r + 8 * g) * 512 + nt * 16 + ln] = acc[r];
  }
  __syncthreads();
  for (int idx = tid; idx < 16 * 128; idx += 256) {
    const int bi = idx >> 7, k = idx & 127;
    const float yt = sYt[bi];
    const float gi = sG[bi * 512 + k] + yt * Wih[k] + bih[k] + bhh[k];
    const float gf = sG[bi * 512 + 128 + k] + yt * Wih[128 + k] + bih[128 + k] + bhh[128 + k];
    const float gg = sG[bi * 512 + 256 + k] + yt * Wih[256 + k] + bih[256 + k] + bhh[256 + k];
    const float go = sG[bi * 512 + 384 + k] + yt * Wih[384 + k] + bih[384 + k] + bhh[384 + k];
    const float co = cS[(b0 + bi) * 128 + k];
    const float cn = sigmoidf_(gf) * co + sigmoidf_(gi) * tanhf(gg);
    const float dn = sigmoidf_(go) * tanhf(cn);
    cS[(b0 + bi) * 128 + k] = cn;
    dS[(b0 + bi) * 128 + k] = dn;
  }
}

// ============ top self-attention (flash over 2048 keys) + residual + head ==
__global__ __launch_bounds__(256) void top_attn_kernel(
    const float* __restrict__ dn, const half_t* __restrict__ Qh,
    const half_t* __restrict__ Kh, const half_t* __restrict__ Vh,
    const float* __restrict__ Wl, const float* __restrict__ bl,
    float* __restrict__ out) {
  __shared__ float sS[64 * 64];
  __shared__ float sO[64 * 128];
  __shared__ float sM[64], sL[64], sR[64];
  const int tid = threadIdx.x, wave = tid >> 5;
  const int lane = tid & 31, ln = lane & 15, g = lane >> 4;
  const int q0 = blockIdx.x * 64;
  const int mtw = wave & 3, dtb = (wave >> 2) * 4;
  v8f acc[4] = {};
  if (tid < 64) { sM[tid] = -1e30f; sL[tid] = 0.f; }
  __syncthreads();
  const float scale = 0.088388347648318447f;  // 1/sqrt(128)
  for (int kb = 0; kb < 32; ++kb) {
    for (int tile = wave; tile < 16; tile += 8) {
      const int mt = tile & 3, kt = tile >> 2;
      v8f s = {};
#pragma unroll
      for (int kc = 0; kc < 128; kc += 32) {
        v16h a = fragA(Qh + (size_t)(q0 + mt * 16) * 128 + kc, 128);
        v16h bb = fragBT(Kh + (size_t)(kb * 64 + kt * 16) * 128 + kc, 128);
        s = wmma32(a, bb, s);
      }
#pragma unroll
      for (int r = 0; r < 8; ++r)
        sS[(mt * 16 + r + 8 * g) * 64 + kt * 16 + ln] = s[r] * scale;
    }
    __syncthreads();
    if (tid < 64) {
      float mx = sM[tid];
      for (int j = 0; j < 64; ++j) mx = fmaxf(mx, sS[tid * 64 + j]);
      const float resc = __expf(sM[tid] - mx);
      float sum = 0.f;
      for (int j = 0; j < 64; ++j) {
        float p = __expf(sS[tid * 64 + j] - mx);
        sS[tid * 64 + j] = p;
        sum += p;
      }
      sL[tid] = sL[tid] * resc + sum;
      sM[tid] = mx;
      sR[tid] = resc;
    }
    __syncthreads();
#pragma unroll
    for (int ti = 0; ti < 4; ++ti) {
      const int dt = dtb + ti;
#pragma unroll
      for (int r = 0; r < 8; ++r) acc[ti][r] *= sR[mtw * 16 + r + 8 * g];
#pragma unroll
      for (int kc = 0; kc < 64; kc += 32) {
        v16h a = fragA(sS + mtw * 16 * 64 + kc, 64);  // f32 -> f16 on load
        v16h bb = fragB(Vh + (size_t)(kb * 64 + kc) * 128 + dt * 16, 128);
        acc[ti] = wmma32(a, bb, acc[ti]);
      }
    }
    __syncthreads();
  }
#pragma unroll
  for (int ti = 0; ti < 4; ++ti) {
    const int dt = dtb + ti;
#pragma unroll
    for (int r = 0; r < 8; ++r) {
      const int row = mtw * 16 + r + 8 * g;
      sO[row * 128 + dt * 16 + ln] = acc[ti][r] / sL[row];
    }
  }
  __syncthreads();
  if (tid < 64) {
    const int row = q0 + tid;
    float val = bl[0];
    for (int j = 0; j < 128; ++j) val += Wl[j] * (sO[tid * 128 + j] + dn[(size_t)row * 128 + j]);
    out[row] = sigmoidf_(val);
  }
}

// ================================ host =====================================
extern "C" void kernel_launch(void* const* d_in, const int* in_sizes, int n_in,
                              void* d_out, int out_size, void* d_ws, size_t ws_size,
                              hipStream_t stream) {
  (void)in_sizes; (void)n_in; (void)out_size; (void)ws_size;
  const int B = 2048;
  const float* decA1W = (const float*)d_in[0];
  const float* decA1b = (const float*)d_in[1];
  const float* decA2W = (const float*)d_in[2];
  const float* decA2b = (const float*)d_in[3];
  const float* decFcW = (const float*)d_in[4];
  const float* decFcb = (const float*)d_in[5];
  const float* decWhh = (const float*)d_in[6];
  const float* decWih = (const float*)d_in[7];
  const float* decBhh = (const float*)d_in[8];
  const float* decBih = (const float*)d_in[9];
  const float* decWk = (const float*)d_in[10];
  const float* decWq = (const float*)d_in[11];
  const float* decWv = (const float*)d_in[12];
  const float* decBk = (const float*)d_in[13];
  const float* decBq = (const float*)d_in[14];
  const float* decBv = (const float*)d_in[15];
  const float* encA1W = (const float*)d_in[16];
  const float* encA1b = (const float*)d_in[17];
  const float* encA2W = (const float*)d_in[18];
  const float* encA2b = (const float*)d_in[19];
  const float* encWhh = (const float*)d_in[20];
  const float* encWih = (const float*)d_in[21];
  const float* encBhh = (const float*)d_in[22];
  const float* encBih = (const float*)d_in[23];
  const float* encWk = (const float*)d_in[24];
  const float* encWq = (const float*)d_in[25];
  const float* encWv = (const float*)d_in[26];
  const float* encBk = (const float*)d_in[27];
  const float* encBq = (const float*)d_in[28];
  const float* encBv = (const float*)d_in[29];
  const float* lastW = (const float*)d_in[30];
  const float* lastB = (const float*)d_in[31];
  const float* topWk = (const float*)d_in[32];
  const float* topWq = (const float*)d_in[33];
  const float* topWv = (const float*)d_in[34];
  const float* topBk = (const float*)d_in[35];
  const float* topBq = (const float*)d_in[36];
  const float* topBv = (const float*)d_in[37];
  const float* x = (const float*)d_in[38];
  const float* y = (const float*)d_in[39];
  float* out = (float*)d_out;

  char* ws = (char*)d_ws;
  size_t off = 0;
  auto alloc = [&](size_t bytes) -> char* {
    char* p = ws + off;
    off += (bytes + 255) & ~(size_t)255;
    return p;
  };
  float* Xsa = (float*)alloc((size_t)B * 64 * 64 * 4);      // 33.5 MB
  float* XsaPart = (float*)alloc((size_t)B * 64 * 64 * 4);  // 33.5 MB
  float* XePart = Xsa;  // reuses Xsa+XsaPart region (67 MB) after encoder
  half_t* Xenc = (half_t*)alloc((size_t)B * 64 * 128 * 2);
  half_t* Xe = (half_t*)alloc((size_t)B * 64 * 128 * 2);
  float* hS = (float*)alloc((size_t)B * 128 * 4);
  float* cS = (float*)alloc((size_t)B * 128 * 4);
  float* dS = (float*)alloc((size_t)B * 128 * 4);
  float* c2S = (float*)alloc((size_t)B * 128 * 4);
  half_t* Qh = (half_t*)alloc((size_t)B * 128 * 2);
  half_t* Kh = (half_t*)alloc((size_t)B * 128 * 2);
  half_t* Vh = (half_t*)alloc((size_t)B * 128 * 2);
  half_t* W1hcP = (half_t*)alloc(256 * 64 * 2);
  half_t* W1xP = (half_t*)alloc(64 * 64 * 2);
  half_t* eWihP = (half_t*)alloc(64 * 512 * 2);
  half_t* eWhhP = (half_t*)alloc(128 * 512 * 2);
  half_t* Wa1dcP = (half_t*)alloc(256 * 128 * 2);
  half_t* Wa1xP = (half_t*)alloc(128 * 128 * 2);
  half_t* dWhhP = (half_t*)alloc(128 * 512 * 2);
  half_t* eQP = (half_t*)alloc(64 * 64 * 2);
  half_t* eKP = (half_t*)alloc(64 * 64 * 2);
  half_t* eVP = (half_t*)alloc(64 * 64 * 2);
  half_t* dQP = (half_t*)alloc(128 * 128 * 2);
  half_t* dKP = (half_t*)alloc(128 * 128 * 2);
  half_t* dVP = (half_t*)alloc(128 * 128 * 2);
  half_t* tQP = (half_t*)alloc(128 * 128 * 2);
  half_t* tKP = (half_t*)alloc(128 * 128 * 2);
  half_t* tVP = (half_t*)alloc(128 * 128 * 2);

  auto cdiv = [](int a, int b) { return (a + b - 1) / b; };
  // states to zero
  zero_kernel<<<cdiv(B * 128, 256), 256, 0, stream>>>(hS, B * 128);
  zero_kernel<<<cdiv(B * 128, 256), 256, 0, stream>>>(cS, B * 128);
  zero_kernel<<<cdiv(B * 128, 256), 256, 0, stream>>>(dS, B * 128);
  zero_kernel<<<cdiv(B * 128, 256), 256, 0, stream>>>(c2S, B * 128);
  // weight packing -> f16 fragment tiles
  tpack_kernel<<<cdiv(256 * 64, 256), 256, 0, stream>>>(encA1W, W1hcP, 256, 64, 320, 0);
  tpack_kernel<<<cdiv(64 * 64, 256), 256, 0, stream>>>(encA1W, W1xP, 64, 64, 320, 256);
  tpack_kernel<<<cdiv(64 * 512, 256), 256, 0, stream>>>(encWih, eWihP, 64, 512, 64, 0);
  tpack_kernel<<<cdiv(128 * 512, 256), 256, 0, stream>>>(encWhh, eWhhP, 128, 512, 128, 0);
  tpack_kernel<<<cdiv(256 * 128, 256), 256, 0, stream>>>(decA1W, Wa1dcP, 256, 128, 384, 0);
  tpack_kernel<<<cdiv(128 * 128, 256), 256, 0, stream>>>(decA1W, Wa1xP, 128, 128, 384, 256);
  tpack_kernel<<<cdiv(128 * 512, 256), 256, 0, stream>>>(decWhh, dWhhP, 128, 512, 128, 0);
  tpack_kernel<<<cdiv(64 * 64, 256), 256, 0, stream>>>(encWq, eQP, 64, 64, 64, 0);
  tpack_kernel<<<cdiv(64 * 64, 256), 256, 0, stream>>>(encWk, eKP, 64, 64, 64, 0);
  tpack_kernel<<<cdiv(64 * 64, 256), 256, 0, stream>>>(encWv, eVP, 64, 64, 64, 0);
  tpack_kernel<<<cdiv(128 * 128, 256), 256, 0, stream>>>(decWq, dQP, 128, 128, 128, 0);
  tpack_kernel<<<cdiv(128 * 128, 256), 256, 0, stream>>>(decWk, dKP, 128, 128, 128, 0);
  tpack_kernel<<<cdiv(128 * 128, 256), 256, 0, stream>>>(decWv, dVP, 128, 128, 128, 0);
  tpack_kernel<<<cdiv(128 * 128, 256), 256, 0, stream>>>(topWq, tQP, 128, 128, 128, 0);
  tpack_kernel<<<cdiv(128 * 128, 256), 256, 0, stream>>>(topWk, tKP, 128, 128, 128, 0);
  tpack_kernel<<<cdiv(128 * 128, 256), 256, 0, stream>>>(topWv, tVP, 128, 128, 128, 0);

  // ---- encoder ----
  enc_sa_kernel<<<B, 256, 0, stream>>>(x, eQP, eKP, eVP, encBq, encBk, encBv, Xsa);
  gemm64_kernel<float><<<B, 256, 0, stream>>>(Xsa, W1xP, encA1b, XsaPart, nullptr, 64, 64);
  for (int t = 0; t < 64; ++t)
    enc_step_kernel<<<B / 16, 256, 0, stream>>>(t, Xsa, XsaPart, W1hcP, encA2W, encA2b,
                                                eWihP, eWhhP, encBih, encBhh, hS, cS, Xenc);
  // ---- decoder ----
  dec_sa_kernel<<<B, 256, 0, stream>>>(Xenc, dQP, dKP, dVP, decBq, decBk, decBv, Xe);
  gemm64_kernel<half_t><<<B, 256, 0, stream>>>(Xe, Wa1xP, decA1b, XePart, nullptr, 128, 128);
  for (int t = 0; t < 64; ++t)
    dec_step_kernel<<<B / 16, 256, 0, stream>>>(t, y, Xe, XePart, Wa1dcP, decA2W, decA2b,
                                                decFcW, decFcb, decWih, dWhhP, decBih,
                                                decBhh, dS, c2S);
  // ---- top attention: QKV projections (dS viewed as 32 row-blocks of 64) ----
  gemm64_kernel<float><<<B / 64, 256, 0, stream>>>(dS, tQP, topBq, nullptr, Qh, 128, 128);
  gemm64_kernel<float><<<B / 64, 256, 0, stream>>>(dS, tKP, topBk, nullptr, Kh, 128, 128);
  gemm64_kernel<float><<<B / 64, 256, 0, stream>>>(dS, tVP, topBv, nullptr, Vh, 128, 128);
  top_attn_kernel<<<B / 64, 256, 0, stream>>>(dS, Qh, Kh, Vh, lastW, lastB, out);
}
